// BackboneCurvatureMixer_80719615361634
// MI455X (gfx1250) — matrix operand order repaired and checked by reading, not compiled, not run on hardware
//
#include <hip/hip_runtime.h>
#include <hip/hip_bf16.h>
#include <math.h>

// ---------------------------------------------------------------------------
// CDNA5 / gfx1250 WMMA types
// ---------------------------------------------------------------------------
typedef __attribute__((ext_vector_type(16))) __bf16 v16bf;
typedef __attribute__((ext_vector_type(8)))  float  v8f;

union FragU {
    uint4          q[2];   // two 16-byte loads
    unsigned short s[16];  // 16 bf16 elements (raw bits)
    v16bf          v;      // WMMA operand
};

// Problem constants
#define NB    16
#define NL    4096
#define ND    1024
#define NR    16
#define NPLU  120
#define NHID  512
#define NTOK  (NB * NL)        // 65536

// float -> bf16 raw bits, round-to-nearest-even
__device__ __forceinline__ unsigned short f2bf(float f) {
    union { float f; unsigned u; } v; v.f = f;
    unsigned u = v.u;
    unsigned r = u + 0x7FFFu + ((u >> 16) & 1u);
    return (unsigned short)(r >> 16);
}

__device__ __forceinline__ float gelu_exact(float x) {
    return 0.5f * x * (1.0f + erff(x * 0.70710678118654752f));
}

__device__ __forceinline__ v8f wmma_bf16(const FragU& a, const FragU& b, v8f c) {
    return __builtin_amdgcn_wmma_f32_16x16x32_bf16(false, a.v, false, b.v,
                                                   (short)0, c, false, false);
}

// ---------------------------------------------------------------------------
// Weight repack: (Kw x Nw) f32 row-major -> bf16 WMMA-B tiles.
// Tile (kt, nt) holds 32 lanes x 16 bf16 contiguous per lane:
//   lane l: N = nt*16 + (l&15), K = kt*32 + (l>=16 ? 8 : 0) + (e&7) + (e>=8 ? 16 : 0),  e = 0..15
// K padded with zeros up to Kpad (multiple of 32).
// ---------------------------------------------------------------------------
__global__ void packw(const float* __restrict__ W, unsigned short* __restrict__ out,
                      int Kw, int Nw, int Kpad) {
    int ntiles = Nw >> 4;
    int total  = (Kpad >> 5) * ntiles * 512;
    int tid = blockIdx.x * blockDim.x + threadIdx.x;
    if (tid >= total) return;
    int tile = tid >> 9;
    int r    = tid & 511;
    int lane = r >> 4;
    int e    = r & 15;
    int kt   = tile / ntiles;
    int nt   = tile - kt * ntiles;
    int K    = kt * 32 + ((lane >> 4) << 3) + (e & 7) + ((e >> 3) << 4);
    int N    = nt * 16 + (lane & 15);
    float v  = (K < Kw) ? W[(size_t)K * Nw + N] : 0.0f;
    out[tid] = f2bf(v);
}

// ---------------------------------------------------------------------------
// z = h @ wred_w + wred_b       (65536 x 1024) @ (1024 x 16)
// One wave per 32-row M-slab: B fragment loaded once per K-step feeds two
// WMMAs; f32->bf16 conversion VALU co-executes with the XDL pipe.
// ---------------------------------------------------------------------------
__global__ __launch_bounds__(256) void zgemm(const float* __restrict__ h,
                                             const unsigned short* __restrict__ wpk,
                                             const float* __restrict__ bias,
                                             float* __restrict__ z) {
    int wave = threadIdx.x >> 5;
    int lane = threadIdx.x & 31;
    int mbase = (blockIdx.x * 8 + wave) * 32;
    int row  = lane & 15;
    int ko   = (lane & 16) ? 8 : 0;   // A-fragment K sub-offset
    int col  = lane & 15;             // C column
    int crow = (lane & 16) ? 8 : 0;   // C row offset

    v8f c0, c1;
    float bv = bias[col];
#pragma unroll
    for (int i = 0; i < 8; ++i) { c0[i] = bv; c1[i] = bv; }

    const float* arow0 = h + (size_t)(mbase + row) * ND;
    const float* arow1 = h + (size_t)(mbase + 16 + row) * ND;
    for (int kt = 0; kt < 32; ++kt) {
        FragU a0, a1, b;
        const unsigned short* bp = wpk + (size_t)kt * 512 + lane * 16;
        b.q[0] = *(const uint4*)bp;
        b.q[1] = *(const uint4*)(bp + 8);
        const float* ap0 = arow0 + kt * 32 + ko;
        const float* ap1 = arow1 + kt * 32 + ko;
#pragma unroll
        for (int i = 0; i < 8; ++i) {
            a0.s[i]     = f2bf(ap0[i]);       // K = ko .. ko+7
            a0.s[8 + i] = f2bf(ap0[16 + i]);  // K = ko+16 .. ko+23
            a1.s[i]     = f2bf(ap1[i]);
            a1.s[8 + i] = f2bf(ap1[16 + i]);
        }
        c0 = wmma_bf16(a0, b, c0);
        c1 = wmma_bf16(a1, b, c1);
    }
#pragma unroll
    for (int i = 0; i < 8; ++i) {
        z[(size_t)(mbase + i + crow) * NR + col]      = c0[i];
        z[(size_t)(mbase + 16 + i + crow) * NR + col] = c1[i];
    }
}

// ---------------------------------------------------------------------------
// Plücker features for deltas {1,2,4}; writes p_bb1 (f32, d_out) and the
// concatenated bf16 GEMM input pcat (NTOK x 384, cols 360..383 zero).
// ---------------------------------------------------------------------------
__global__ __launch_bounds__(256) void plucker(const float* __restrict__ z,
                                               const unsigned char* __restrict__ mask,
                                               float* __restrict__ p_bb1,
                                               unsigned short* __restrict__ pcat) {
    int t = blockIdx.x * blockDim.x + threadIdx.x;   // token index, < NTOK
    int il = t & (NL - 1);                           // position within sequence
    float u[NR];
#pragma unroll
    for (int i = 0; i < NR; ++i) u[i] = z[(size_t)t * NR + i];
    float mi = mask[t] ? 1.0f : 0.0f;

    const int deltas[3] = {1, 2, 4};
#pragma unroll
    for (int d = 0; d < 3; ++d) {
        int delta = deltas[d];
        bool inr = (il + delta) < NL;
        float v[NR];
#pragma unroll
        for (int i = 0; i < NR; ++i)
            v[i] = inr ? z[(size_t)(t + delta) * NR + i] : 0.0f;
        float mj = inr ? (mask[t + delta] ? 1.0f : 0.0f) : 0.0f;
        float mm = mi * mj;

        float pv[NPLU];
        float s = 0.0f;
        int k = 0;
#pragma unroll
        for (int i = 0; i < NR; ++i) {
#pragma unroll
            for (int j = i + 1; j < NR; ++j) {
                float x = u[i] * v[j] - u[j] * v[i];
                pv[k] = x;
                s += x * x;
                ++k;
            }
        }
        float n = sqrtf(fmaxf(s, 1e-16f));
        float inv = mm / fmaxf(n, 1e-8f);
#pragma unroll
        for (int kk = 0; kk < NPLU; ++kk) {
            float o = pv[kk] * inv;
            if (d == 0) p_bb1[(size_t)t * NPLU + kk] = o;
            pcat[(size_t)t * 384 + d * NPLU + kk] = f2bf(o);
        }
    }
#pragma unroll
    for (int cpd = 360; cpd < 384; ++cpd) pcat[(size_t)t * 384 + cpd] = 0;
}

// ---------------------------------------------------------------------------
// kappa = (p_fwd - 2p + p_bwd) * mask ; f32 to d_out, bf16 (padded to 128) to ws
// ---------------------------------------------------------------------------
__global__ __launch_bounds__(256) void kappak(const float* __restrict__ p,
                                              const unsigned char* __restrict__ mask,
                                              float* __restrict__ kout,
                                              unsigned short* __restrict__ kws) {
    int idx = blockIdx.x * blockDim.x + threadIdx.x;  // NTOK * 128
    int t = idx >> 7;
    int c = idx & 127;
    if (c >= NPLU) { kws[(size_t)t * 128 + c] = 0; return; }
    int il = t & (NL - 1);
    float pc = p[(size_t)t * NPLU + c];
    float pf = (il < NL - 1) ? p[(size_t)(t + 1) * NPLU + c] : 0.0f;
    float pb = (il > 0)      ? p[(size_t)(t - 1) * NPLU + c] : 0.0f;
    float m  = mask[t] ? 1.0f : 0.0f;
    float kv = (pf - 2.0f * pc + pb) * m;
    kout[(size_t)t * NPLU + c] = kv;
    kws[(size_t)t * 128 + c]   = f2bf(kv);
}

// ---------------------------------------------------------------------------
// Fused 2-layer MLP:  out = GELU(act @ W1 + b1) @ W2 + b2
// act: bf16 row-major (NTOK x Kpad); W1/W2 packed bf16 B-tiles.
// 2 waves/block, M=32 rows per wave (two 16-row tiles), 4 N-tiles per
// accumulation group -> 8 independent WMMAs per K-step with shared A frags.
// Hidden staged as bf16 in LDS (32 KB/wave, 64 KB/block).
// ---------------------------------------------------------------------------
__global__ __launch_bounds__(64) void mlp2(const unsigned short* __restrict__ act,
                                           int ktiles1,
                                           const unsigned short* __restrict__ w1pk,
                                           const float* __restrict__ b1,
                                           const unsigned short* __restrict__ w2pk,
                                           const float* __restrict__ b2,
                                           float* __restrict__ out) {
    __shared__ unsigned short hid[2 * 32 * NHID];   // 64 KB: 32x512 bf16 per wave
    int wave = threadIdx.x >> 5;
    int lane = threadIdx.x & 31;
    int mbase = (blockIdx.x * 2 + wave) * 32;
    int row  = lane & 15;
    int ko   = (lane & 16) ? 8 : 0;
    int col  = lane & 15;
    int crow = (lane & 16) ? 8 : 0;
    unsigned short* myh = hid + wave * 32 * NHID;
    int Kpad = ktiles1 << 5;

    const unsigned short* arow0 = act + (size_t)(mbase + row) * Kpad;
    const unsigned short* arow1 = act + (size_t)(mbase + 16 + row) * Kpad;

    // ---- stage 1: hidden = GELU(act @ W1 + b1), bf16 -> LDS ----
    for (int ntg = 0; ntg < 8; ++ntg) {            // 8 groups x 4 N-tiles = 512
        v8f c0[4], c1[4];
#pragma unroll
        for (int n4 = 0; n4 < 4; ++n4) {
            float bv = b1[(ntg * 4 + n4) * 16 + col];
#pragma unroll
            for (int i = 0; i < 8; ++i) { c0[n4][i] = bv; c1[n4][i] = bv; }
        }
        for (int kt = 0; kt < ktiles1; ++kt) {
            FragU a0, a1;
            const unsigned short* ap0 = arow0 + kt * 32 + ko;
            const unsigned short* ap1 = arow1 + kt * 32 + ko;
            a0.q[0] = *(const uint4*)ap0;
            a0.q[1] = *(const uint4*)(ap0 + 16);
            a1.q[0] = *(const uint4*)ap1;
            a1.q[1] = *(const uint4*)(ap1 + 16);
#pragma unroll
            for (int n4 = 0; n4 < 4; ++n4) {
                int nt = ntg * 4 + n4;
                FragU b;
                const unsigned short* bp = w1pk + ((size_t)kt * 32 + nt) * 512 + lane * 16;
                b.q[0] = *(const uint4*)bp;
                b.q[1] = *(const uint4*)(bp + 8);
                c0[n4] = wmma_bf16(a0, b, c0[n4]);
                c1[n4] = wmma_bf16(a1, b, c1[n4]);
            }
        }
#pragma unroll
        for (int n4 = 0; n4 < 4; ++n4) {
            int nc = (ntg * 4 + n4) * 16 + col;
#pragma unroll
            for (int i = 0; i < 8; ++i) {
                myh[(i + crow) * NHID + nc]      = f2bf(gelu_exact(c0[n4][i]));
                myh[(16 + i + crow) * NHID + nc] = f2bf(gelu_exact(c1[n4][i]));
            }
        }
    }

    // ---- stage 2: out = hidden @ W2 + b2 ----
    for (int ntg = 0; ntg < 16; ++ntg) {           // 16 groups x 4 N-tiles = 1024
        v8f c0[4], c1[4];
#pragma unroll
        for (int n4 = 0; n4 < 4; ++n4) {
            float bv = b2[(ntg * 4 + n4) * 16 + col];
#pragma unroll
            for (int i = 0; i < 8; ++i) { c0[n4][i] = bv; c1[n4][i] = bv; }
        }
        for (int kt = 0; kt < 16; ++kt) {
            FragU a0, a1;
            const unsigned short* ap0 = myh + row * NHID + kt * 32 + ko;
            const unsigned short* ap1 = myh + (16 + row) * NHID + kt * 32 + ko;
            a0.q[0] = *(const uint4*)ap0;                     // ds_load_b128
            a0.q[1] = *(const uint4*)(ap0 + 16);
            a1.q[0] = *(const uint4*)ap1;
            a1.q[1] = *(const uint4*)(ap1 + 16);
#pragma unroll
            for (int n4 = 0; n4 < 4; ++n4) {
                int nt = ntg * 4 + n4;
                FragU b;
                const unsigned short* bp = w2pk + ((size_t)kt * 64 + nt) * 512 + lane * 16;
                b.q[0] = *(const uint4*)bp;
                b.q[1] = *(const uint4*)(bp + 8);
                __builtin_prefetch((const void*)(bp + 512 * 64), 0, 1); // next K tile
                c0[n4] = wmma_bf16(a0, b, c0[n4]);
                c1[n4] = wmma_bf16(a1, b, c1[n4]);
            }
        }
#pragma unroll
        for (int n4 = 0; n4 < 4; ++n4) {
            int nc = (ntg * 4 + n4) * 16 + col;
#pragma unroll
            for (int i = 0; i < 8; ++i) {
                out[(size_t)(mbase + i + crow) * ND + nc]      = c0[n4][i];
                out[(size_t)(mbase + 16 + i + crow) * ND + nc] = c1[n4][i];
            }
        }
    }
}

// ---------------------------------------------------------------------------
// launch
// ---------------------------------------------------------------------------
extern "C" void kernel_launch(void* const* d_in, const int* in_sizes, int n_in,
                              void* d_out, int out_size, void* d_ws, size_t ws_size,
                              hipStream_t stream) {
    const float*         h      = (const float*)d_in[0];
    const unsigned char* mask   = (const unsigned char*)d_in[1];
    const float*         wred_w = (const float*)d_in[2];
    const float*         wred_b = (const float*)d_in[3];
    const float*         bb_w1  = (const float*)d_in[4];
    const float*         bb_b1  = (const float*)d_in[5];
    const float*         bb_w2  = (const float*)d_in[6];
    const float*         bb_b2  = (const float*)d_in[7];
    const float*         cv_w1  = (const float*)d_in[8];
    const float*         cv_b1  = (const float*)d_in[9];
    const float*         cv_w2  = (const float*)d_in[10];
    const float*         cv_b2  = (const float*)d_in[11];

    float* out    = (float*)d_out;
    float* z      = out;
    float* g_bb   = out + (size_t)NTOK * NR;
    float* g_curv = g_bb + (size_t)NTOK * ND;
    float* p_bb1  = g_curv + (size_t)NTOK * ND;
    float* kappa  = p_bb1 + (size_t)NTOK * NPLU;

    // workspace layout (bytes, all 1 KB-aligned)
    char* ws = (char*)d_ws;
    unsigned short* wredPk = (unsigned short*)(ws + 0);          //   32 KB
    unsigned short* bbW1Pk = (unsigned short*)(ws + 32768);      //  384 KB
    unsigned short* bbW2Pk = (unsigned short*)(ws + 425984);     // 1024 KB
    unsigned short* cvW1Pk = (unsigned short*)(ws + 1474560);    //  128 KB
    unsigned short* cvW2Pk = (unsigned short*)(ws + 1605632);    // 1024 KB
    unsigned short* pcat   = (unsigned short*)(ws + 2654208);    //   48 MB
    unsigned short* kws    = (unsigned short*)(ws + 52985856);   //   16 MB

    // 1) repack weights to bf16 WMMA-B tiles
    {
        int n;
        n = (1024 / 32) * (16 / 16) * 512;
        packw<<<(n + 255) / 256, 256, 0, stream>>>(wred_w, wredPk, 1024, 16, 1024);
        n = (384 / 32) * (512 / 16) * 512;
        packw<<<(n + 255) / 256, 256, 0, stream>>>(bb_w1, bbW1Pk, 360, 512, 384);
        n = (512 / 32) * (1024 / 16) * 512;
        packw<<<(n + 255) / 256, 256, 0, stream>>>(bb_w2, bbW2Pk, 512, 1024, 512);
        n = (128 / 32) * (512 / 16) * 512;
        packw<<<(n + 255) / 256, 256, 0, stream>>>(cv_w1, cvW1Pk, 120, 512, 128);
        n = (512 / 32) * (1024 / 16) * 512;
        packw<<<(n + 255) / 256, 256, 0, stream>>>(cv_w2, cvW2Pk, 512, 1024, 512);
    }

    // 2) z = h @ wred_w + b          (32 rows/wave, 8 waves/block)
    zgemm<<<NTOK / (32 * 8), 256, 0, stream>>>(h, wredPk, wred_b, z);

    // 3) Plücker features + pcat (bf16)
    plucker<<<NTOK / 256, 256, 0, stream>>>(z, mask, p_bb1, pcat);

    // 4) kappa + bf16 copy
    kappak<<<(NTOK * 128) / 256, 256, 0, stream>>>(p_bb1, mask, kappa, kws);

    // 5) g_bb   = GELU(pcat  @ bb_w1 + b1) @ bb_w2 + b2
    mlp2<<<NTOK / (32 * 2), 64, 0, stream>>>(pcat, 12, bbW1Pk, bb_b1, bbW2Pk, bb_b2, g_bb);

    // 6) g_curv = GELU(kappa @ cv_w1 + b1) @ cv_w2 + b2
    mlp2<<<NTOK / (32 * 2), 64, 0, stream>>>(kws, 4, cvW1Pk, cv_b1, cvW2Pk, cv_b2, g_curv);
}
